// MHA_65738769433305
// MI455X (gfx1250) — matrix operand order repaired and checked by reading, not compiled, not run on hardware
//
#include <hip/hip_runtime.h>
#include <hip/hip_bf16.h>

typedef _Float16 half_t;
typedef __attribute__((ext_vector_type(8)))  _Float16 v8h;
typedef __attribute__((ext_vector_type(16))) _Float16 v16h;
typedef __attribute__((ext_vector_type(8)))  float    v8f;
typedef __attribute__((ext_vector_type(4)))  int      v4i;

#define DIM   1024
#define NH    16
#define HD    64
#define SEQ   2048
#define BATCH 4
#define ROWS  (BATCH * SEQ)   // 8192

// ---------------------------------------------------------------------------
// CDNA5 async global->LDS copy (ASYNCcnt-tracked), with sync fallback
// ---------------------------------------------------------------------------
#if defined(__AMDGCN__) && defined(__gfx1250__) && \
    __has_builtin(__builtin_amdgcn_global_load_async_to_lds_b128) && \
    __has_builtin(__builtin_amdgcn_s_wait_asynccnt)
#define USE_ASYNC_LDS 1
#else
#define USE_ASYNC_LDS 0
#endif

#if USE_ASYNC_LDS
typedef __attribute__((address_space(1))) v4i* gas_v4i;
typedef __attribute__((address_space(3))) v4i* las_v4i;
__device__ __forceinline__ void cp16(half_t* lds_dst, const half_t* gsrc) {
    __builtin_amdgcn_global_load_async_to_lds_b128(
        (gas_v4i)gsrc, (las_v4i)lds_dst, 0, 0);
}
__device__ __forceinline__ void wait_async() {
    __builtin_amdgcn_s_wait_asynccnt(0);
}
#else
__device__ __forceinline__ void cp16(half_t* lds_dst, const half_t* gsrc) {
    *(v8h*)lds_dst = *(const v8h*)gsrc;
}
__device__ __forceinline__ void wait_async() {}
#endif

__device__ inline v16h pack16(v8h lo, v8h hi) {
    v16h r;
#pragma unroll
    for (int i = 0; i < 8; ++i) { r[i] = lo[i]; r[8 + i] = hi[i]; }
    return r;
}

// ---------------------------------------------------------------------------
// 1) fp32 -> fp16 elementwise convert
// ---------------------------------------------------------------------------
__global__ __launch_bounds__(512) void cvt_f16_kernel(const float* __restrict__ in,
                                                      half_t* __restrict__ out, int n) {
    int i = blockIdx.x * blockDim.x + threadIdx.x;
    if (i < n) out[i] = (half_t)in[i];
}

// ---------------------------------------------------------------------------
// 2) fp32 [R][C] -> fp16 transposed [C][R]   (R, C multiples of 32)
// ---------------------------------------------------------------------------
__global__ __launch_bounds__(256) void transpose_cvt_kernel(const float* __restrict__ in,
                                                            half_t* __restrict__ out,
                                                            int R, int C) {
    __shared__ float tile[32][33];
    int bx = blockIdx.x * 32;   // along C
    int by = blockIdx.y * 32;   // along R
    int tx = threadIdx.x;       // 0..31
    int ty = threadIdx.y;       // 0..7
#pragma unroll
    for (int i = 0; i < 32; i += 8)
        tile[ty + i][tx] = in[(size_t)(by + ty + i) * C + bx + tx];
    __syncthreads();
#pragma unroll
    for (int i = 0; i < 32; i += 8)
        out[(size_t)(bx + ty + i) * R + by + tx] = (half_t)tile[tx][ty + i];
}

// ---------------------------------------------------------------------------
// 3) QKV projection: Xh[8192x1024] @ WqkvT[3072x1024] with async double-buffered
//    LDS pipeline; results scattered into Q (pre-scaled), K, V^T f16 buffers.
// ---------------------------------------------------------------------------
#define AS_STRIDE 40
#define BS_STRIDE 40

__global__ __launch_bounds__(128) void qkv_gemm_kernel(const half_t* __restrict__ Xh,
                                                       const half_t* __restrict__ WT,
                                                       half_t* __restrict__ Qb,
                                                       half_t* __restrict__ Kb,
                                                       half_t* __restrict__ Vt) {
    __shared__ __align__(16) half_t As[2][128 * AS_STRIDE];
    __shared__ __align__(16) half_t Bs[2][64 * BS_STRIDE];

    const int tid  = threadIdx.x;
    const int wave = tid >> 5;
    const int lane = tid & 31;
    const int l16  = lane & 15;
    const int hi   = lane >> 4;
    const int M0   = blockIdx.x * 128;
    const int N0   = blockIdx.y * 64;

    const half_t* srcA0 = Xh + (size_t)(M0 + tid) * DIM;
    const int rB = tid >> 1;
    const int cB = (tid & 1) * 16;
    const half_t* srcB0 = WT + (size_t)(N0 + rB) * DIM + cB;

    auto load_tiles = [&](int kb, int buf) {
        const half_t* srcA = srcA0 + kb;
        half_t* dstA = As[buf] + tid * AS_STRIDE;
        cp16(dstA,      srcA);
        cp16(dstA + 8,  srcA + 8);
        cp16(dstA + 16, srcA + 16);
        cp16(dstA + 24, srcA + 24);
        const half_t* srcB = srcB0 + kb;
        half_t* dstB = Bs[buf] + rB * BS_STRIDE + cB;
        cp16(dstB,     srcB);
        cp16(dstB + 8, srcB + 8);
#if !USE_ASYNC_LDS
        __builtin_prefetch(srcA + 32, 0, 1);
        __builtin_prefetch(srcB + 32, 0, 1);
#endif
    };

    v8f acc[2][4] = {};

    load_tiles(0, 0);
    wait_async();
    __syncthreads();

    int buf = 0;
    for (int kb = 0; kb < DIM; kb += 32) {
        if (kb + 32 < DIM) load_tiles(kb + 32, buf ^ 1);

        v16h afr[2];
#pragma unroll
        for (int m = 0; m < 2; ++m) {
            const half_t* p = As[buf] + (wave * 32 + m * 16 + l16) * AS_STRIDE + hi * 8;
            afr[m] = pack16(*(const v8h*)p, *(const v8h*)(p + 16));
        }
#pragma unroll
        for (int n = 0; n < 4; ++n) {
            const half_t* p = Bs[buf] + (n * 16 + l16) * BS_STRIDE + hi * 8;
            v16h bfr = pack16(*(const v8h*)p, *(const v8h*)(p + 16));
#pragma unroll
            for (int m = 0; m < 2; ++m)
                acc[m][n] = __builtin_amdgcn_wmma_f32_16x16x32_f16(
                    false, afr[m], false, bfr, (short)0, acc[m][n], false, false);
        }

        wait_async();
        __syncthreads();
        buf ^= 1;
    }

    // scatter: section 0 = Q (scale 1/8), 1 = K, 2 = V (transposed)
    const int sec = N0 >> 10;
    const int h   = (N0 & 1023) >> 6;
#pragma unroll
    for (int m = 0; m < 2; ++m) {
#pragma unroll
        for (int n = 0; n < 4; ++n) {
#pragma unroll
            for (int r = 0; r < 8; ++r) {
                int rowg = M0 + wave * 32 + m * 16 + r + hi * 8;
                int b = rowg >> 11, s = rowg & 2047;
                int bh = b * NH + h;
                int hc = n * 16 + l16;
                float v = acc[m][n][r];
                if (sec == 0)
                    Qb[((size_t)bh * SEQ + s) * HD + hc] = (half_t)(v * 0.125f);
                else if (sec == 1)
                    Kb[((size_t)bh * SEQ + s) * HD + hc] = (half_t)v;
                else
                    Vt[((size_t)bh * HD + hc) * SEQ + s] = (half_t)v;
            }
        }
    }
}

// ---------------------------------------------------------------------------
// 4) Flash attention. WG = one (b,h) x 64 query rows, 4 waves x 16 rows.
//    KV blocks of 64 keys; K / V^T async double-buffered in LDS.
// ---------------------------------------------------------------------------
#define KS_STRIDE 72   // 64 + 8 pad halves, 144 B (16B-aligned rows)

__global__ __launch_bounds__(128) void attn_kernel(const half_t* __restrict__ Qb,
                                                   const half_t* __restrict__ Kb,
                                                   const half_t* __restrict__ Vt,
                                                   half_t* __restrict__ Ctx) {
    __shared__ __align__(16) half_t Ks[2][64 * KS_STRIDE];
    __shared__ __align__(16) half_t Vs[2][64 * KS_STRIDE];
    __shared__ __align__(16) half_t Ps[4 * 16 * KS_STRIDE];

    const int tid  = threadIdx.x;
    const int wave = tid >> 5;
    const int lane = tid & 31;
    const int l16  = lane & 15;
    const int hi   = lane >> 4;
    const int qt   = blockIdx.x;          // query tile (64 rows)
    const int bh   = blockIdx.y;          // (b*NH + h)

    const half_t* Qp = Qb + (size_t)bh * SEQ * HD;
    const half_t* Kp = Kb + (size_t)bh * SEQ * HD;
    const half_t* Vp = Vt + (size_t)bh * HD * SEQ;

    const int qbase = qt * 64 + wave * 16;

    const int rL = tid >> 1;
    const int cL = (tid & 1) * 32;

    auto load_kv = [&](int j, int buf) {
        int kv0 = j * 64;
        const half_t* ks = Kp + (size_t)(kv0 + rL) * HD + cL;
        half_t* kd = Ks[buf] + rL * KS_STRIDE + cL;
        cp16(kd,      ks);
        cp16(kd + 8,  ks + 8);
        cp16(kd + 16, ks + 16);
        cp16(kd + 24, ks + 24);
        const half_t* vsrc = Vp + (size_t)rL * SEQ + kv0 + cL;
        half_t* vd = Vs[buf] + rL * KS_STRIDE + cL;
        cp16(vd,      vsrc);
        cp16(vd + 8,  vsrc + 8);
        cp16(vd + 16, vsrc + 16);
        cp16(vd + 24, vsrc + 24);
    };

    // Q fragments (rows fixed); Q is pre-scaled by 1/sqrt(hd)
    v16h qfr[2];
    {
        const half_t* qrow = Qp + (size_t)(qbase + l16) * HD;
#pragma unroll
        for (int kk = 0; kk < 2; ++kk)
            qfr[kk] = pack16(*(const v8h*)(qrow + kk * 32 + hi * 8),
                             *(const v8h*)(qrow + kk * 32 + 16 + hi * 8));
    }

    v8f   acc[4] = {};
    float mrow[8], lrow[8];
#pragma unroll
    for (int r = 0; r < 8; ++r) { mrow[r] = -1e30f; lrow[r] = 0.0f; }

    half_t* Pw = Ps + wave * 16 * KS_STRIDE;

    load_kv(0, 0);
    wait_async();
    __syncthreads();

    int buf = 0;
    for (int j = 0; j <= qt; ++j) {
        const int kv0 = j * 64;
        if (j < qt) load_kv(j + 1, buf ^ 1);

        // scores: 16 queries x 64 keys
        v8f s[4] = {};
#pragma unroll
        for (int n = 0; n < 4; ++n) {
#pragma unroll
            for (int kk = 0; kk < 2; ++kk) {
                const half_t* p = Ks[buf] + (n * 16 + l16) * KS_STRIDE + kk * 32 + hi * 8;
                v16h bfr = pack16(*(const v8h*)p, *(const v8h*)(p + 16));
                s[n] = __builtin_amdgcn_wmma_f32_16x16x32_f16(
                    false, qfr[kk], false, bfr, (short)0, s[n], false, false);
            }
        }

        if (j == qt) {   // diagonal block: causal mask
#pragma unroll
            for (int n = 0; n < 4; ++n)
#pragma unroll
                for (int r = 0; r < 8; ++r) {
                    int key = kv0 + n * 16 + l16;
                    int qrw = qbase + r + hi * 8;
                    if (key > qrw) s[n][r] = -1e30f;
                }
        }

        // per-row block max (reduce across 16-lane column group)
        float bm[8];
#pragma unroll
        for (int r = 0; r < 8; ++r) {
            float v = fmaxf(fmaxf(s[0][r], s[1][r]), fmaxf(s[2][r], s[3][r]));
#pragma unroll
            for (int d = 1; d < 16; d <<= 1) v = fmaxf(v, __shfl_xor(v, d));
            bm[r] = v;
        }

        float alpha[8], rs[8];
#pragma unroll
        for (int r = 0; r < 8; ++r) {
            float mnew = fmaxf(mrow[r], bm[r]);
            alpha[r] = __expf(mrow[r] - mnew);
            mrow[r]  = mnew;
            rs[r]    = 0.0f;
        }
#pragma unroll
        for (int n = 0; n < 4; ++n)
#pragma unroll
            for (int r = 0; r < 8; ++r) {
                float p = __expf(s[n][r] - mrow[r]);
                s[n][r] = p;
                rs[r] += p;
            }
#pragma unroll
        for (int r = 0; r < 8; ++r) {
#pragma unroll
            for (int d = 1; d < 16; d <<= 1) rs[r] += __shfl_xor(rs[r], d);
            lrow[r] = lrow[r] * alpha[r] + rs[r];
        }
#pragma unroll
        for (int n = 0; n < 4; ++n)
#pragma unroll
            for (int r = 0; r < 8; ++r) acc[n][r] *= alpha[r];

        // transpose P (C-layout -> A-layout) through wave-private LDS
#pragma unroll
        for (int n = 0; n < 4; ++n)
#pragma unroll
            for (int r = 0; r < 8; ++r)
                Pw[(r + hi * 8) * KS_STRIDE + n * 16 + l16] = (half_t)s[n][r];

        // ctx += P @ Vblock   (M=16 queries, K=64 keys, N=64 hd)
#pragma unroll
        for (int kk = 0; kk < 2; ++kk) {
            const half_t* pp = Pw + l16 * KS_STRIDE + kk * 32 + hi * 8;
            v16h pa = pack16(*(const v8h*)pp, *(const v8h*)(pp + 16));
#pragma unroll
            for (int n = 0; n < 4; ++n) {
                const half_t* vp2 = Vs[buf] + (n * 16 + l16) * KS_STRIDE + kk * 32 + hi * 8;
                v16h bfr = pack16(*(const v8h*)vp2, *(const v8h*)(vp2 + 16));
                acc[n] = __builtin_amdgcn_wmma_f32_16x16x32_f16(
                    false, pa, false, bfr, (short)0, acc[n], false, false);
            }
        }

        if (j < qt) wait_async();
        __syncthreads();
        buf ^= 1;
    }

    // finalize: ctx = acc / l, stored f16 into [B*S][DIM] with head offset
    const int b = bh >> 4, h = bh & 15;
#pragma unroll
    for (int n = 0; n < 4; ++n)
#pragma unroll
        for (int r = 0; r < 8; ++r) {
            int srow = qbase + r + hi * 8;
            size_t rowg = (size_t)(b * SEQ + srow);
            Ctx[rowg * DIM + h * HD + n * 16 + l16] = (half_t)(acc[n][r] / lrow[r]);
        }
}

// ---------------------------------------------------------------------------
// 5) Output projection: ctx[8192x1024](f16) @ WoutT[1024x1024](f16) + bias -> f32
// ---------------------------------------------------------------------------
__global__ __launch_bounds__(128) void out_gemm_kernel(const half_t* __restrict__ Ch,
                                                       const half_t* __restrict__ WT,
                                                       const float* __restrict__ bias,
                                                       float* __restrict__ Out) {
    __shared__ __align__(16) half_t As[2][128 * AS_STRIDE];
    __shared__ __align__(16) half_t Bs[2][64 * BS_STRIDE];

    const int tid  = threadIdx.x;
    const int wave = tid >> 5;
    const int lane = tid & 31;
    const int l16  = lane & 15;
    const int hi   = lane >> 4;
    const int M0   = blockIdx.x * 128;
    const int N0   = blockIdx.y * 64;

    const half_t* srcA0 = Ch + (size_t)(M0 + tid) * DIM;
    const int rB = tid >> 1;
    const int cB = (tid & 1) * 16;
    const half_t* srcB0 = WT + (size_t)(N0 + rB) * DIM + cB;

    auto load_tiles = [&](int kb, int buf) {
        const half_t* srcA = srcA0 + kb;
        half_t* dstA = As[buf] + tid * AS_STRIDE;
        cp16(dstA,      srcA);
        cp16(dstA + 8,  srcA + 8);
        cp16(dstA + 16, srcA + 16);
        cp16(dstA + 24, srcA + 24);
        const half_t* srcB = srcB0 + kb;
        half_t* dstB = Bs[buf] + rB * BS_STRIDE + cB;
        cp16(dstB,     srcB);
        cp16(dstB + 8, srcB + 8);
#if !USE_ASYNC_LDS
        __builtin_prefetch(srcA + 32, 0, 1);
        __builtin_prefetch(srcB + 32, 0, 1);
#endif
    };

    v8f acc[2][4] = {};

    load_tiles(0, 0);
    wait_async();
    __syncthreads();

    int buf = 0;
    for (int kb = 0; kb < DIM; kb += 32) {
        if (kb + 32 < DIM) load_tiles(kb + 32, buf ^ 1);

        v16h afr[2];
#pragma unroll
        for (int m = 0; m < 2; ++m) {
            const half_t* p = As[buf] + (wave * 32 + m * 16 + l16) * AS_STRIDE + hi * 8;
            afr[m] = pack16(*(const v8h*)p, *(const v8h*)(p + 16));
        }
#pragma unroll
        for (int n = 0; n < 4; ++n) {
            const half_t* p = Bs[buf] + (n * 16 + l16) * BS_STRIDE + hi * 8;
            v16h bfr = pack16(*(const v8h*)p, *(const v8h*)(p + 16));
#pragma unroll
            for (int m = 0; m < 2; ++m)
                acc[m][n] = __builtin_amdgcn_wmma_f32_16x16x32_f16(
                    false, afr[m], false, bfr, (short)0, acc[m][n], false, false);
        }

        wait_async();
        __syncthreads();
        buf ^= 1;
    }

    float bv[4];
#pragma unroll
    for (int n = 0; n < 4; ++n) bv[n] = bias[N0 + n * 16 + l16];

#pragma unroll
    for (int m = 0; m < 2; ++m)
#pragma unroll
        for (int n = 0; n < 4; ++n)
#pragma unroll
            for (int r = 0; r < 8; ++r) {
                int rowg = M0 + wave * 32 + m * 16 + r + hi * 8;
                Out[(size_t)rowg * DIM + N0 + n * 16 + l16] = acc[m][n][r] + bv[n];
            }
}

// ---------------------------------------------------------------------------
// launch
// ---------------------------------------------------------------------------
extern "C" void kernel_launch(void* const* d_in, const int* in_sizes, int n_in,
                              void* d_out, int out_size, void* d_ws, size_t ws_size,
                              hipStream_t stream) {
    (void)in_sizes; (void)n_in; (void)out_size; (void)ws_size;
    const float* X    = (const float*)d_in[0];
    const float* Wqkv = (const float*)d_in[1];
    const float* Wout = (const float*)d_in[2];
    const float* bout = (const float*)d_in[3];
    float* out = (float*)d_out;

    char* ws = (char*)d_ws;
    half_t* Xh    = (half_t*)(ws);                       // 16 MB
    half_t* WqkvT = (half_t*)(ws + (size_t)16777216);    //  6 MB
    half_t* WoutT = (half_t*)(ws + (size_t)23068672);    //  2 MB
    half_t* Qb    = (half_t*)(ws + (size_t)25165824);    // 16 MB
    half_t* Kb    = (half_t*)(ws + (size_t)41943040);    // 16 MB
    half_t* Vtb   = (half_t*)(ws + (size_t)58720256);    // 16 MB
    half_t* Ctx   = (half_t*)(ws + (size_t)75497472);    // 16 MB

    cvt_f16_kernel<<<(ROWS * DIM) / 512, 512, 0, stream>>>(X, Xh, ROWS * DIM);
    transpose_cvt_kernel<<<dim3(3 * DIM / 32, DIM / 32), dim3(32, 8), 0, stream>>>(
        Wqkv, WqkvT, DIM, 3 * DIM);
    transpose_cvt_kernel<<<dim3(DIM / 32, DIM / 32), dim3(32, 8), 0, stream>>>(
        Wout, WoutT, DIM, DIM);
    qkv_gemm_kernel<<<dim3(ROWS / 128, 3 * DIM / 64), 128, 0, stream>>>(
        Xh, WqkvT, Qb, Kb, Vtb);
    attn_kernel<<<dim3(SEQ / 64, BATCH * NH), 128, 0, stream>>>(Qb, Kb, Vtb, Ctx);
    out_gemm_kernel<<<dim3(ROWS / 128, DIM / 64), 128, 0, stream>>>(
        Ctx, WoutT, bout, out);
}
